// SafeMambaBlock_4827543241088
// MI455X (gfx1250) — matrix operand order repaired and checked by reading, not compiled, not run on hardware
//
#include <hip/hip_runtime.h>

typedef _Float16 h16;
typedef __attribute__((ext_vector_type(16))) _Float16 v16h;
typedef __attribute__((ext_vector_type(8)))  float    v8f;

#define B_SZ   2
#define L_SZ   2048
#define DM     1024
#define DI     2048
#define NSTATE 16
#define DTR    64
#define XDBL   (DTR + 2*NSTATE)   /* 96 */
#define MROWS  (B_SZ*L_SZ)        /* 4096 */

#define BM  128
#define BN  128
#define BKK 32
#define LDPAD 8
#define LDW (BKK + LDPAD)         /* 40 halves = 80 B row stride */

#if defined(__has_builtin)
#if __has_builtin(__builtin_amdgcn_tensor_load_to_lds)
#define HAVE_TDM 1
#endif
#endif

__device__ __forceinline__ float siluf(float x) { return x / (1.f + __expf(-x)); }

#ifdef HAVE_TDM
typedef unsigned int u32x4 __attribute__((ext_vector_type(4)));
typedef int          i32x4 __attribute__((ext_vector_type(4)));
typedef int          i32x8 __attribute__((ext_vector_type(8)));
typedef __attribute__((address_space(3))) _Float16 lds_h16;

// flat address of an LDS object: addr[31:0] is the wave-relative LDS byte offset
__device__ __forceinline__ unsigned lds_off(const void* p) {
  return (unsigned)(unsigned long long)p;
}

// TDM descriptor (ISA cdna5 ch8): load a (rows x 32) f16 tile from a row-major
// tensor (row stride = ld elements) into LDS. rem_k/rem_rows give remaining
// tensor extent from the tile origin so OOB rows/cols are hardware zero-filled.
// pad_interval=16 DWORDs (one 32-half row), pad_amount=4 DWORDs -> reproduces
// the +8-half LDS row padding (row pitch 80 B) in the DMA itself.
__device__ __forceinline__ void tdm_load_tile_f16(const h16* gsrc, unsigned lds_byte,
                                                  unsigned rem_k, unsigned rem_rows,
                                                  unsigned ld) {
  unsigned long long ga = (unsigned long long)gsrc;
  u32x4 g0;
  g0[0] = 1u;                                   // count=1, user desc, no gather
  g0[1] = lds_byte;                             // lds_addr
  g0[2] = (unsigned)ga;                         // global_addr[31:0]
  g0[3] = (unsigned)((ga >> 32) & 0x01FFFFFFu)  // global_addr[56:32]
        | (2u << 30);                           // type = 2 ("image")
  i32x8 g1;
  g1[0] = (int)((1u << 16)                      // data_size = 2 bytes
              | (1u << 20)                      // pad_enable
              | (3u << 22)                      // pad_interval: 16 DWORDs
              | (3u << 25));                    // pad_amount: 4 DWORDs
  g1[1] = (int)((rem_k & 0xFFFFu) << 16);                               // tensor_dim0[15:0]
  g1[2] = (int)(((rem_k >> 16) & 0xFFFFu) | ((rem_rows & 0xFFFFu) << 16)); // dim0 hi | dim1 lo
  g1[3] = (int)(((rem_rows >> 16) & 0xFFFFu) | ((unsigned)BKK << 16));  // dim1 hi | tile_dim0=32
  g1[4] = (int)BM;                              // tile_dim1 = 128 rows, tile_dim2 = 0
  g1[5] = (int)ld;                              // tensor_dim0_stride[31:0]
  g1[6] = 0;                                    // stride hi, dim1_stride lo
  g1[7] = 0;
  i32x4 gz = {0, 0, 0, 0};
#if __clang_major__ >= 23
  i32x8 gz8 = {0, 0, 0, 0, 0, 0, 0, 0};
  __builtin_amdgcn_tensor_load_to_lds(g0, g1, gz, gz, gz8, 0);
#else
  __builtin_amdgcn_tensor_load_to_lds(g0, g1, gz, gz, 0);
#endif
}
#endif  // HAVE_TDM

// ---------------------------------------------------------------------------
// LayerNorm (f32 in) -> f16 normalized activations
// ---------------------------------------------------------------------------
__global__ __launch_bounds__(256)
void ln_f16_kernel(const float* __restrict__ x, const float* __restrict__ g,
                   const float* __restrict__ bb, h16* __restrict__ out) {
  __shared__ float red[8];
  int row = blockIdx.x;
  int tid = threadIdx.x;
  const float* xr = x + (size_t)row * DM;
  float v[4];
  float s = 0.f;
#pragma unroll
  for (int k = 0; k < 4; ++k) { v[k] = xr[tid + 256 * k]; s += v[k]; }
#pragma unroll
  for (int o = 16; o > 0; o >>= 1) s += __shfl_xor(s, o, 32);
  if ((tid & 31) == 0) red[tid >> 5] = s;
  __syncthreads();
  if (tid == 0) {
    float t = 0.f;
#pragma unroll
    for (int i = 0; i < 8; ++i) t += red[i];
    red[0] = t;
  }
  __syncthreads();
  float mu = red[0] * (1.f / DM);
  __syncthreads();
  float sv = 0.f;
#pragma unroll
  for (int k = 0; k < 4; ++k) { float d = v[k] - mu; sv += d * d; }
#pragma unroll
  for (int o = 16; o > 0; o >>= 1) sv += __shfl_xor(sv, o, 32);
  if ((tid & 31) == 0) red[tid >> 5] = sv;
  __syncthreads();
  if (tid == 0) {
    float t = 0.f;
#pragma unroll
    for (int i = 0; i < 8; ++i) t += red[i];
    red[0] = t;
  }
  __syncthreads();
  float rinv = rsqrtf(red[0] * (1.f / DM) + 1e-5f);
  h16* orow = out + (size_t)row * DM;
#pragma unroll
  for (int k = 0; k < 4; ++k) {
    int c = tid + 256 * k;
    orow[c] = (h16)((v[k] - mu) * rinv * g[c] + bb[c]);
  }
}

// ---------------------------------------------------------------------------
__global__ void cvt_f32_f16_kernel(const float* __restrict__ s, h16* __restrict__ d, int n) {
  int i = blockIdx.x * blockDim.x + threadIdx.x;
  if (i < n) d[i] = (h16)s[i];
}

// ---------------------------------------------------------------------------
// Tiled WMMA GEMM: C[M,N] = A[M,K] * Bw[N,K]^T (f16 in, f32 acc)
// 256 thr = 8 wave32; 128x128 C tile; 32x64 per wave (2x4 x v_wmma 16x16x32)
// TDM path: wave0 drives double-buffered TENSOR_LOAD_TO_LDS of A/B tiles,
// s_wait_tensorcnt pipelines tile t+1 DMA under tile t WMMAs.
// ---------------------------------------------------------------------------
template <bool OUT_HALF, bool RESID>
__global__ __launch_bounds__(256)
void gemm_wmma_kernel(const h16* __restrict__ A, const h16* __restrict__ Bw,
                      float* __restrict__ Cf, h16* __restrict__ Ch,
                      const float* __restrict__ resid, int M, int N, int K) {
  __shared__ __align__(16) h16 As[2][BM][LDW];
  __shared__ __align__(16) h16 Bs[2][BN][LDW];

  int tid = threadIdx.x;
  int lane = tid & 31;
  int wave = tid >> 5;   // 0..7
  int waveM = wave & 3;  // 4 groups of 32 rows
  int waveN = wave >> 2; // 2 groups of 64 cols
  int lr = lane & 15;
  int hi = lane >> 4;
  int m0 = blockIdx.y * BM;
  int n0 = blockIdx.x * BN;

  v8f acc[2][4];
  const v8f vzero = {0.f, 0.f, 0.f, 0.f, 0.f, 0.f, 0.f, 0.f};
#pragma unroll
  for (int i = 0; i < 2; ++i)
#pragma unroll
    for (int j = 0; j < 4; ++j) acc[i][j] = vzero;

  int ntiles = K / BKK;

#ifdef HAVE_TDM
  unsigned asBase = lds_off(&As[0][0][0]);
  unsigned bsBase = lds_off(&Bs[0][0][0]);
  // launder so the compiler cannot assume the TDM never wrote these buffers
  asm volatile("" : "+v"(asBase), "+v"(bsBase));
  lds_h16* Asl = (lds_h16*)asBase;
  lds_h16* Bsl = (lds_h16*)bsBase;

  unsigned remA = (unsigned)(M - m0);
  unsigned remB = (unsigned)(N - n0);
  const h16* Abase = A + (size_t)m0 * K;
  const h16* Bbase = Bw + (size_t)n0 * K;

  if (tid < 32) {  // wave 0 drives the DMA
    tdm_load_tile_f16(Abase, lds_off(&As[0][0][0]), (unsigned)K, remA, (unsigned)K);
    tdm_load_tile_f16(Bbase, lds_off(&Bs[0][0][0]), (unsigned)K, remB, (unsigned)K);
  }

  for (int t = 0; t < ntiles; ++t) {
    int cur = t & 1;
    __syncthreads();  // buffer cur^1 is free (consumed at iteration t-1)
    if (tid < 32) {
      if (t + 1 < ntiles) {
        int k1 = (t + 1) * BKK;
        tdm_load_tile_f16(Abase + k1, lds_off(&As[cur ^ 1][0][0]),
                          (unsigned)(K - k1), remA, (unsigned)K);
        tdm_load_tile_f16(Bbase + k1, lds_off(&Bs[cur ^ 1][0][0]),
                          (unsigned)(K - k1), remB, (unsigned)K);
        __builtin_amdgcn_s_wait_tensorcnt(2);  // tile t pair done, t+1 in flight
      } else {
        __builtin_amdgcn_s_wait_tensorcnt(0);
      }
    }
    __syncthreads();  // tile t visible to all 8 waves

    lds_h16* Acur = Asl + (size_t)cur * BM * LDW;
    lds_h16* Bcur = Bsl + (size_t)cur * BN * LDW;
    v16h af[2], bf[4];
#pragma unroll
    for (int mi = 0; mi < 2; ++mi) {
      int ar = waveM * 32 + mi * 16 + lr;
      int ko = hi * 8;
      lds_h16* ap = Acur + ar * LDW;
#pragma unroll
      for (int j = 0; j < 8; ++j) af[mi][j] = ap[ko + j];
#pragma unroll
      for (int j = 0; j < 8; ++j) af[mi][8 + j] = ap[16 + ko + j];
    }
#pragma unroll
    for (int ni = 0; ni < 4; ++ni) {
      int bc = waveN * 64 + ni * 16 + lr;
      int kb = hi * 8;
      lds_h16* bp = Bcur + bc * LDW;
#pragma unroll
      for (int j = 0; j < 16; ++j) bf[ni][j] = bp[(j < 8) ? (kb + j) : (16 + kb + j - 8)];
    }
#pragma unroll
    for (int mi = 0; mi < 2; ++mi)
#pragma unroll
      for (int ni = 0; ni < 4; ++ni)
        acc[mi][ni] = __builtin_amdgcn_wmma_f32_16x16x32_f16(
            false, af[mi], false, bf[ni], (short)0, acc[mi][ni], false, false);
  }
#else  // ---------------- fallback: manual staging -----
  for (int t = 0; t < ntiles; ++t) {
    int k0 = t * BKK;
#pragma unroll
    for (int it = 0; it < 2; ++it) {
      int chunk = tid + it * 256;
      int r = chunk >> 2;
      int cs = (chunk & 3) * 8;
      const h16* asrc = A + (size_t)(m0 + r) * K + k0 + cs;
      if (k0 + BKK < K) __builtin_prefetch(asrc + BKK, 0, 3);
      *(uint4*)(&As[0][r][cs]) = *(const uint4*)asrc;
      uint4 bz = {0u, 0u, 0u, 0u};
      int nIdx = n0 + r;
      if (nIdx < N) {
        const h16* bsrc = Bw + (size_t)nIdx * K + k0 + cs;
        bz = *(const uint4*)bsrc;
      }
      *(uint4*)(&Bs[0][r][cs]) = bz;
    }
    __syncthreads();
    v16h af[2], bf[4];
#pragma unroll
    for (int mi = 0; mi < 2; ++mi) {
      int ar = waveM * 32 + mi * 16 + lr;
      int ko = hi * 8;
#pragma unroll
      for (int j = 0; j < 8; ++j) af[mi][j] = As[0][ar][ko + j];
#pragma unroll
      for (int j = 0; j < 8; ++j) af[mi][8 + j] = As[0][ar][16 + ko + j];
    }
#pragma unroll
    for (int ni = 0; ni < 4; ++ni) {
      int bc = waveN * 64 + ni * 16 + lr;
      int kb = hi * 8;
#pragma unroll
      for (int j = 0; j < 16; ++j) bf[ni][j] = Bs[0][bc][(j < 8) ? (kb + j) : (16 + kb + j - 8)];
    }
#pragma unroll
    for (int mi = 0; mi < 2; ++mi)
#pragma unroll
      for (int ni = 0; ni < 4; ++ni)
        acc[mi][ni] = __builtin_amdgcn_wmma_f32_16x16x32_f16(
            false, af[mi], false, bf[ni], (short)0, acc[mi][ni], false, false);
    __syncthreads();
  }
#endif

  // epilogue: f32 C layout — lane lr = N col, vgpr r: M = hi*8 + r
#pragma unroll
  for (int mi = 0; mi < 2; ++mi) {
#pragma unroll
    for (int ni = 0; ni < 4; ++ni) {
      int col = n0 + waveN * 64 + ni * 16 + lr;
      if (col < N) {
#pragma unroll
        for (int r8 = 0; r8 < 8; ++r8) {
          int row = m0 + waveM * 32 + mi * 16 + hi * 8 + r8;
          float v = acc[mi][ni][r8];
          if (RESID) v += resid[(size_t)row * N + col];
          if (OUT_HALF) Ch[(size_t)row * N + col] = (h16)v;
          else          Cf[(size_t)row * N + col] = v;
        }
      }
    }
  }
}

// ---------------------------------------------------------------------------
// depthwise causal conv1d (k=4) + SiLU, reads xi half of xz (cols [0,DI))
// ---------------------------------------------------------------------------
__global__ __launch_bounds__(256)
void conv_silu_kernel(const h16* __restrict__ xz16, const float* __restrict__ cw,
                      const float* __restrict__ cb, h16* __restrict__ xi16) {
  int idx = blockIdx.x * blockDim.x + threadIdx.x;
  if (idx >= MROWS * DI) return;
  int row = idx / DI;
  int d = idx - row * DI;
  int b = row / L_SZ;
  int l = row - b * L_SZ;
  float acc = cb[d];
#pragma unroll
  for (int j = 0; j < 4; ++j) {
    int ll = l - 3 + j;
    if (ll >= 0)
      acc += cw[d * 4 + j] * (float)xz16[(size_t)(b * L_SZ + ll) * (2 * DI) + d];
  }
  xi16[(size_t)row * DI + d] = (h16)siluf(acc);
}

// ---------------------------------------------------------------------------
__global__ void extract_dt_kernel(const float* __restrict__ xdbl, h16* __restrict__ dt16) {
  int idx = blockIdx.x * blockDim.x + threadIdx.x;
  if (idx >= MROWS * DTR) return;
  int row = idx / DTR;
  int c = idx - row * DTR;
  dt16[idx] = (h16)xdbl[(size_t)row * XDBL + c];
}

// ---------------------------------------------------------------------------
// selective scan + softplus + SiLU(z) gate; one thread per (batch,channel)
// ---------------------------------------------------------------------------
__global__ __launch_bounds__(256)
void scan_kernel(const h16* __restrict__ xz16, const h16* __restrict__ xi16,
                 const float* __restrict__ dtraw, const float* __restrict__ xdbl,
                 const float* __restrict__ A_log, const float* __restrict__ Dp,
                 const float* __restrict__ dtb, h16* __restrict__ y16) {
  int g = blockIdx.x * blockDim.x + threadIdx.x;
  if (g >= B_SZ * DI) return;
  int b = g / DI;
  int d = g - b * DI;

  float Av[NSTATE], h[NSTATE];
#pragma unroll
  for (int n = 0; n < NSTATE; ++n) { Av[n] = -__expf(A_log[d * NSTATE + n]); h[n] = 0.f; }
  float Dd = Dp[d];
  float bias = dtb[d];

  for (int l = 0; l < L_SZ; ++l) {
    size_t row = (size_t)b * L_SZ + l;
    float u = (float)xi16[row * DI + d];
    float xv = dtraw[row * DI + d] + bias;
    float dt = (xv > 20.f) ? xv : log1pf(__expf(xv));
    const float* bc = xdbl + row * XDBL;
    float y = 0.f;
#pragma unroll
    for (int n = 0; n < NSTATE; ++n) {
      float dA = __expf(dt * Av[n]);
      h[n] = dA * h[n] + dt * bc[DTR + n] * u;
      y += h[n] * bc[DTR + NSTATE + n];
    }
    y += u * Dd;
    float z = (float)xz16[row * (2 * DI) + DI + d];
    y *= siluf(z);
    y16[row * DI + d] = (h16)y;
  }
}

// ---------------------------------------------------------------------------
extern "C" void kernel_launch(void* const* d_in, const int* in_sizes, int n_in,
                              void* d_out, int out_size, void* d_ws, size_t ws_size,
                              hipStream_t stream) {
  const float* x     = (const float*)d_in[0];
  const float* inW   = (const float*)d_in[1];
  const float* convW = (const float*)d_in[2];
  const float* convB = (const float*)d_in[3];
  const float* xpW   = (const float*)d_in[4];
  const float* dtW   = (const float*)d_in[5];
  const float* dtB   = (const float*)d_in[6];
  const float* A_log = (const float*)d_in[7];
  const float* Dp    = (const float*)d_in[8];
  const float* outW  = (const float*)d_in[9];
  const float* lnG   = (const float*)d_in[10];
  const float* lnB   = (const float*)d_in[11];
  float* out = (float*)d_out;

  char* w = (char*)d_ws;
  size_t off = 0;
  auto alloc = [&](size_t bytes) -> void* {
    void* p = w + off;
    off = (off + bytes + 255) & ~(size_t)255;
    return p;
  };
  h16*   xn16  = (h16*)alloc((size_t)MROWS * DM * 2);
  h16*   w_in  = (h16*)alloc((size_t)2 * DI * DM * 2);
  h16*   w_xp  = (h16*)alloc((size_t)XDBL * DI * 2);
  h16*   w_dt  = (h16*)alloc((size_t)DI * DTR * 2);
  h16*   w_out = (h16*)alloc((size_t)DM * DI * 2);
  h16*   xz16  = (h16*)alloc((size_t)MROWS * 2 * DI * 2);
  h16*   xi16  = (h16*)alloc((size_t)MROWS * DI * 2);
  float* xdbl  = (float*)alloc((size_t)MROWS * XDBL * 4);
  h16*   dt16  = (h16*)alloc((size_t)MROWS * DTR * 2);
  float* dtraw = (float*)alloc((size_t)MROWS * DI * 4);
  h16*   y16   = (h16*)alloc((size_t)MROWS * DI * 2);

  {
    int n1 = 2 * DI * DM;
    cvt_f32_f16_kernel<<<(n1 + 255) / 256, 256, 0, stream>>>(inW, w_in, n1);
    int n2 = XDBL * DI;
    cvt_f32_f16_kernel<<<(n2 + 255) / 256, 256, 0, stream>>>(xpW, w_xp, n2);
    int n3 = DI * DTR;
    cvt_f32_f16_kernel<<<(n3 + 255) / 256, 256, 0, stream>>>(dtW, w_dt, n3);
    int n4 = DM * DI;
    cvt_f32_f16_kernel<<<(n4 + 255) / 256, 256, 0, stream>>>(outW, w_out, n4);
  }

  ln_f16_kernel<<<MROWS, 256, 0, stream>>>(x, lnG, lnB, xn16);

  gemm_wmma_kernel<true, false><<<dim3((2 * DI) / BN, MROWS / BM), 256, 0, stream>>>(
      xn16, w_in, nullptr, xz16, nullptr, MROWS, 2 * DI, DM);

  conv_silu_kernel<<<(MROWS * DI) / 256, 256, 0, stream>>>(xz16, convW, convB, xi16);

  gemm_wmma_kernel<false, false><<<dim3((XDBL + BN - 1) / BN, MROWS / BM), 256, 0, stream>>>(
      xi16, w_xp, xdbl, nullptr, nullptr, MROWS, XDBL, DI);

  extract_dt_kernel<<<(MROWS * DTR) / 256, 256, 0, stream>>>(xdbl, dt16);

  gemm_wmma_kernel<false, false><<<dim3(DI / BN, MROWS / BM), 256, 0, stream>>>(
      dt16, w_dt, dtraw, nullptr, nullptr, MROWS, DI, DTR);

  scan_kernel<<<(B_SZ * DI) / 256, 256, 0, stream>>>(xz16, xi16, dtraw, xdbl,
                                                     A_log, Dp, dtB, y16);

  gemm_wmma_kernel<false, true><<<dim3(DM / BN, MROWS / BM), 256, 0, stream>>>(
      y16, w_out, out, nullptr, x, MROWS, DM, DI);
}
